// DeepDownstreamFork_64407329571549
// MI455X (gfx1250) — matrix operand order repaired and checked by reading, not compiled, not run on hardware
//
#include <hip/hip_runtime.h>
#include <stdint.h>

// ---------------------------------------------------------------------------
// Feature probe: gfx1250 async global->LDS copies (ASYNCcnt-tracked)
// ---------------------------------------------------------------------------
#if defined(__has_builtin)
#  if __has_builtin(__builtin_amdgcn_global_load_async_to_lds_b128) && \
      __has_builtin(__builtin_amdgcn_s_wait_asynccnt)
#    define HAVE_ASYNC_LDS 1
#  endif
#endif
#ifndef HAVE_ASYNC_LDS
#  define HAVE_ASYNC_LDS 0
#endif

// ---------------------------------------------------------------------------
// Types
// ---------------------------------------------------------------------------
typedef __attribute__((ext_vector_type(16))) __bf16    v16bf;
typedef __attribute__((ext_vector_type(8)))  float     v8f;
typedef __attribute__((ext_vector_type(4)))  unsigned  u32x4;
typedef __attribute__((ext_vector_type(2)))  unsigned  u32x2;
typedef __attribute__((ext_vector_type(4)))  float     f32x4;

// Matches the builtin's parameter type: 'int __attribute__((vector_size(16)))'
typedef int gvec4i __attribute__((vector_size(16)));
#if HAVE_ASYNC_LDS
typedef __attribute__((address_space(1))) gvec4i* glob_v4i_p;
typedef __attribute__((address_space(3))) gvec4i* lds_v4i_p;
#endif

struct Frag { u32x4 lo, hi; };   // 32 bytes == v16bf

__device__ __forceinline__ unsigned short f2bf(float f) {
  union { float f; unsigned u; } v; v.f = f;
  unsigned r = v.u + 0x7FFFu + ((v.u >> 16) & 1u);   // round-to-nearest-even
  return (unsigned short)(r >> 16);
}

// ---------------------------------------------------------------------------
// f32 -> bf16 bulk convert (x activations)
// ---------------------------------------------------------------------------
__global__ __launch_bounds__(256) void cvt_f32_bf16(const float* __restrict__ in,
                                                    unsigned short* __restrict__ out) {
  size_t i = ((size_t)blockIdx.x * 256 + threadIdx.x) * 4;
  f32x4 v = *(const f32x4*)(in + i);
  u32x2 p;
  p.x = (unsigned)f2bf(v.x) | ((unsigned)f2bf(v.y) << 16);
  p.y = (unsigned)f2bf(v.z) | ((unsigned)f2bf(v.w) << 16);
  *(u32x2*)(out + i) = p;
}

// ---------------------------------------------------------------------------
// Dequantize weight [N][K] with per-128x128 block scale (scale==nullptr -> 1.0)
// grid: (K/1024, N)
// ---------------------------------------------------------------------------
__global__ __launch_bounds__(256) void dequant_bf16(const float* __restrict__ W,
                                                    const float* __restrict__ scale,
                                                    unsigned short* __restrict__ out,
                                                    int K, int scaleCols) {
  int n = blockIdx.y;
  int k = (blockIdx.x * 256 + threadIdx.x) * 4;
  size_t idx = (size_t)n * K + k;
  f32x4 v = *(const f32x4*)(W + idx);
  float sc = scale ? scale[(n >> 7) * scaleCols + (k >> 7)] : 1.0f;
  u32x2 p;
  p.x = (unsigned)f2bf(v.x * sc) | ((unsigned)f2bf(v.y * sc) << 16);
  p.y = (unsigned)f2bf(v.z * sc) | ((unsigned)f2bf(v.w * sc) << 16);
  *(u32x2*)(out + idx) = p;
}

// ---------------------------------------------------------------------------
// Wsum = deq(W_s1,s1) + deq(W_s2,s2) -> bf16, both [1024][1024]. grid (1,1024)
// ---------------------------------------------------------------------------
__global__ __launch_bounds__(256) void wsum_bf16(const float* __restrict__ W1,
                                                 const float* __restrict__ s1,
                                                 const float* __restrict__ W2,
                                                 const float* __restrict__ s2,
                                                 unsigned short* __restrict__ out) {
  int n = blockIdx.y;
  int k = (blockIdx.x * 256 + threadIdx.x) * 4;
  size_t idx = (size_t)n * 1024 + k;
  f32x4 a = *(const f32x4*)(W1 + idx);
  f32x4 b = *(const f32x4*)(W2 + idx);
  float sa = s1[(n >> 7) * 8 + (k >> 7)];
  float sb = s2[(n >> 7) * 8 + (k >> 7)];
  u32x2 p;
  p.x = (unsigned)f2bf(a.x * sa + b.x * sb) | ((unsigned)f2bf(a.y * sa + b.y * sb) << 16);
  p.y = (unsigned)f2bf(a.z * sa + b.z * sb) | ((unsigned)f2bf(a.w * sa + b.w * sb) << 16);
  *(u32x2*)(out + idx) = p;
}

// ---------------------------------------------------------------------------
// Tiled bf16 GEMM:  C[M][N] = A[M][K] @ B[N][K]^T    (both row-major, bf16)
// Double-buffered LDS; async global->LDS when available (ASYNCcnt), else
// global_load_b128 + ds_store_b128 fallback.
// Supports K-concatenated second pair (A1,B1,K1) for the fused GEMM3 + s-path.
// EPI==0: relu -> bf16 store ; EPI==1: fp32 store
// block = 256 threads (8 wave32), tile 128x128, K-step 32.
// ---------------------------------------------------------------------------
#define LDSW 40   // padded LDS row stride in bf16 elems (80B, 16B aligned)

template <int EPI>
__global__ __launch_bounds__(256) void gemm_bf16(
    const unsigned short* __restrict__ A0, const unsigned short* __restrict__ B0, int K0,
    const unsigned short* __restrict__ A1, const unsigned short* __restrict__ B1, int K1,
    unsigned short* __restrict__ Ob, float* __restrict__ Of, int N) {
  __shared__ __align__(16) unsigned short sA[2][128 * LDSW];
  __shared__ __align__(16) unsigned short sB[2][128 * LDSW];

  const int tid   = threadIdx.x;
  const int wave  = tid >> 5;
  const int lane  = tid & 31;
  const int mBase = blockIdx.y * 128;
  const int nBase = blockIdx.x * 128;

  const int r  = lane & 15;            // row within 16-tile
  const int kh = (lane >> 4) * 8;      // A-fragment K split (ISA 16-bit A layout)
  const int kb = (lane >> 4) * 8;      // B-fragment K split (ISA 16-bit layout, same as A)

  v8f acc[8] = {};

  // Stage 128x32 A/B tiles for K-step `sidx` into LDS buffer `buf`.
  auto stage = [&](int sidx, int buf) {
    int kk = sidx << 5;
    const unsigned short* Ap;
    const unsigned short* Bp;
    int ld;
    if (kk < K0) { Ap = A0; Bp = B0; ld = K0; }
    else         { Ap = A1; Bp = B1; ld = K1; kk -= K0; }
    for (int it = 0; it < 2; ++it) {
      int c   = tid + it * 256;        // 0..511 chunk id (16B each)
      int rr  = c >> 2;
      int col = (c & 3) << 3;
      const unsigned short* ga = Ap + (size_t)(mBase + rr) * ld + kk + col;
      const unsigned short* gb = Bp + (size_t)(nBase + rr) * ld + kk + col;
      unsigned short* la = &sA[buf][rr * LDSW + col];
      unsigned short* lb = &sB[buf][rr * LDSW + col];
#if HAVE_ASYNC_LDS
      __builtin_amdgcn_global_load_async_to_lds_b128(
          (glob_v4i_p)ga, (lds_v4i_p)la, 0, 0);
      __builtin_amdgcn_global_load_async_to_lds_b128(
          (glob_v4i_p)gb, (lds_v4i_p)lb, 0, 0);
#else
      *(u32x4*)la = *(const u32x4*)ga;
      *(u32x4*)lb = *(const u32x4*)gb;
#endif
    }
  };

  const int steps = (K0 + K1) >> 5;

  // Prologue: fill buffer 0
  stage(0, 0);
#if HAVE_ASYNC_LDS
  __builtin_amdgcn_s_wait_asynccnt(0);
#endif
  __syncthreads();

  for (int s = 0; s < steps; ++s) {
    const int cur = s & 1;
    // Prefetch next tile into the back buffer while we compute from `cur`.
    if (s + 1 < steps) stage(s + 1, cur ^ 1);

    // A fragment: this wave's 16 rows, reused across all 8 column tiles
    Frag fa;
    const unsigned short* pa = &sA[cur][(wave * 16 + r) * LDSW + kh];
    fa.lo = *(const u32x4*)pa;
    fa.hi = *(const u32x4*)(pa + 16);
    v16bf va = __builtin_bit_cast(v16bf, fa);

#pragma unroll
    for (int j = 0; j < 8; ++j) {
      Frag fb;
      const unsigned short* pb = &sB[cur][(j * 16 + r) * LDSW + kb];
      fb.lo = *(const u32x4*)pb;
      fb.hi = *(const u32x4*)(pb + 16);
      v16bf vb = __builtin_bit_cast(v16bf, fb);
      acc[j] = __builtin_amdgcn_wmma_f32_16x16x32_bf16(
          false, va, false, vb, (short)0, acc[j], false, false);
    }

#if HAVE_ASYNC_LDS
    __builtin_amdgcn_s_wait_asynccnt(0);
#endif
    __syncthreads();   // back buffer fully written; front buffer done being read
  }

  // Epilogue: C/D layout -> lane (l&15)=col, vgpr v + 8*(l>>4) = row
  const int row0 = mBase + wave * 16 + ((lane >> 4) << 3);
  const int col0 = nBase + (lane & 15);
#pragma unroll
  for (int j = 0; j < 8; ++j) {
    int col = col0 + j * 16;
#pragma unroll
    for (int v = 0; v < 8; ++v) {
      size_t idx = (size_t)(row0 + v) * (size_t)N + col;
      float val = acc[j][v];
      if (EPI == 0) {
        Ob[idx] = f2bf(val > 0.0f ? val : 0.0f);   // relu + bf16
      } else {
        Of[idx] = val;                              // fp32 (pre-layernorm)
      }
    }
  }
}

// ---------------------------------------------------------------------------
// In-place LayerNorm over rows of 1024 fp32. grid = M rows, 256 threads.
// ---------------------------------------------------------------------------
__global__ __launch_bounds__(256) void ln_inplace(float* __restrict__ y,
                                                  const float* __restrict__ gamma,
                                                  const float* __restrict__ beta) {
  __shared__ float sSum[256];
  __shared__ float sSq[256];
  const int tid = threadIdx.x;
  float* row = y + (size_t)blockIdx.x * 1024;

  f32x4 v = *(const f32x4*)(row + tid * 4);
  sSum[tid] = v.x + v.y + v.z + v.w;
  sSq[tid]  = v.x * v.x + v.y * v.y + v.z * v.z + v.w * v.w;
  __syncthreads();
  for (int off = 128; off > 0; off >>= 1) {
    if (tid < off) { sSum[tid] += sSum[tid + off]; sSq[tid] += sSq[tid + off]; }
    __syncthreads();
  }
  float mu   = sSum[0] * (1.0f / 1024.0f);
  float var  = sSq[0] * (1.0f / 1024.0f) - mu * mu;
  float rstd = rsqrtf(var + 1e-5f);

  f32x4 g = *(const f32x4*)(gamma + tid * 4);
  f32x4 b = *(const f32x4*)(beta + tid * 4);
  f32x4 o;
  o.x = (v.x - mu) * rstd * g.x + b.x;
  o.y = (v.y - mu) * rstd * g.y + b.y;
  o.z = (v.z - mu) * rstd * g.z + b.z;
  o.w = (v.w - mu) * rstd * g.w + b.w;
  *(f32x4*)(row + tid * 4) = o;
}

// ---------------------------------------------------------------------------
// Launch
// ---------------------------------------------------------------------------
extern "C" void kernel_launch(void* const* d_in, const int* in_sizes, int n_in,
                              void* d_out, int out_size, void* d_ws, size_t ws_size,
                              hipStream_t stream) {
  const float* x       = (const float*)d_in[0];   // [8,4096,1024]
  const float* W_large = (const float*)d_in[1];   // [4096,1024]
  const float* W_s1    = (const float*)d_in[2];   // [1024,1024]
  const float* W_s2    = (const float*)d_in[3];   // [1024,1024]
  const float* W_c1    = (const float*)d_in[4];   // [4096,4096]
  const float* W_c2    = (const float*)d_in[5];   // [1024,4096]
  const float* gamma   = (const float*)d_in[6];   // [1024]
  const float* beta    = (const float*)d_in[7];   // [1024]
  const float* s_large = (const float*)d_in[8];   // [32,8]
  const float* s_s1    = (const float*)d_in[9];   // [8,8]
  const float* s_s2    = (const float*)d_in[10];  // [8,8]
  float* out = (float*)d_out;                     // [8,4096,1024] fp32

  const int M = 32768, H = 1024, L = 4096;

  char* ws = (char*)d_ws;
  unsigned short* xb   = (unsigned short*)(ws);                 //  64 MiB  x bf16
  unsigned short* Wl   = (unsigned short*)(ws + 67108864);      //   8 MiB  deq W_large
  unsigned short* Wc1  = (unsigned short*)(ws + 75497472);      //  32 MiB  W_c1 bf16
  unsigned short* Wc2  = (unsigned short*)(ws + 109051904);     //   8 MiB  W_c2 bf16
  unsigned short* Wsm  = (unsigned short*)(ws + 117440512);     //   2 MiB  deq(Ws1)+deq(Ws2)
  unsigned short* big1 = (unsigned short*)(ws + 119537664);     // 256 MiB  relu(x@Wl^T)
  unsigned short* big2 = (unsigned short*)(ws + 387973120);     // 256 MiB  relu(big1@Wc1^T)

  // --- precision / scale folding passes (cheap: weights + x once) ---
  cvt_f32_bf16<<<dim3((M * H) / 1024), 256, 0, stream>>>(x, xb);
  dequant_bf16<<<dim3(1, L), 256, 0, stream>>>(W_large, s_large, Wl, H, H / 128);
  dequant_bf16<<<dim3(4, L), 256, 0, stream>>>(W_c1, nullptr, Wc1, L, 0);
  dequant_bf16<<<dim3(4, H), 256, 0, stream>>>(W_c2, nullptr, Wc2, L, 0);
  wsum_bf16<<<dim3(1, H), 256, 0, stream>>>(W_s1, s_s1, W_s2, s_s2, Wsm);

  // --- GEMM chain (all WMMA bf16, 128x128 tiles, double-buffered LDS) ---
  // G1: big1 = relu(x @ Wl^T)            [32768,4096]
  gemm_bf16<0><<<dim3(L / 128, M / 128), 256, 0, stream>>>(
      xb, Wl, H, nullptr, nullptr, 0, big1, nullptr, L);
  // G2: big2 = relu(big1 @ Wc1^T)        [32768,4096]
  gemm_bf16<0><<<dim3(L / 128, M / 128), 256, 0, stream>>>(
      big1, Wc1, L, nullptr, nullptr, 0, big2, nullptr, L);
  // G3 (fused): out = big2 @ Wc2^T + x @ Wsum^T   (K = 4096 + 1024)
  gemm_bf16<1><<<dim3(H / 128, M / 128), 256, 0, stream>>>(
      big2, Wc2, L, xb, Wsm, H, nullptr, out, H);

  // --- LayerNorm in place on d_out ---
  ln_inplace<<<dim3(M), 256, 0, stream>>>(out, gamma, beta);
}
